// FlexAttnFunc_386547057220
// MI455X (gfx1250) — matrix operand order repaired and checked by reading, not compiled, not run on hardware
//
#include <hip/hip_runtime.h>
#include <hip/hip_bf16.h>

// ---------------------------------------------------------------------------
// Flash-attention for the flex-attn reference on gfx1250 (MI455X).
//   SEQ = 4128 (= 258*16 = 129*32), N_HEADS = 12, HEAD_DIM = 128.
//   Math path: bf16 WMMA (v_wmma_f32_16x16x32_bf16), f32 online softmax.
//   Pre-passes: K fp32[s][h][d] -> bf16 Kb[h][s][d]   (row-major, B-frag ready)
//               V fp32[s][h][d] -> bf16 Vt[h][d][s]   (transposed, B-frag ready)
// ---------------------------------------------------------------------------

#define SEQ 4128
#define NH  12
#define HD  128

typedef __attribute__((ext_vector_type(16))) __bf16 v16bf;
typedef __attribute__((ext_vector_type(8)))  float  v8f;

static __device__ __forceinline__ v8f wmma_bf16(v16bf a, v16bf b, v8f c) {
  // D = A(16x32 bf16) x B(32x16 bf16) + C(16x16 f32)
  return __builtin_amdgcn_wmma_f32_16x16x32_bf16(
      /*neg_a=*/false, a, /*neg_b=*/false, b,
      /*c_mod=*/(short)0, c, /*reuse_a=*/false, /*reuse_b=*/false);
}

static __device__ __forceinline__ unsigned short f2bf_bits(float f) {
  return __builtin_bit_cast(unsigned short, (__bf16)f);
}

// 8 floats from p0 -> elems 0..7, 8 floats from p1 -> elems 8..15 (Q A-frag)
static __device__ __forceinline__ v16bf load8p8(const float* p0, const float* p1) {
  const float4* f0 = (const float4*)p0;
  const float4* f1 = (const float4*)p1;
  float4 a = f0[0], b = f0[1], c = f1[0], d = f1[1];
  v16bf r;
  r[0]=(__bf16)a.x;  r[1]=(__bf16)a.y;  r[2]=(__bf16)a.z;  r[3]=(__bf16)a.w;
  r[4]=(__bf16)b.x;  r[5]=(__bf16)b.y;  r[6]=(__bf16)b.z;  r[7]=(__bf16)b.w;
  r[8]=(__bf16)c.x;  r[9]=(__bf16)c.y;  r[10]=(__bf16)c.z; r[11]=(__bf16)c.w;
  r[12]=(__bf16)d.x; r[13]=(__bf16)d.y; r[14]=(__bf16)d.z; r[15]=(__bf16)d.w;
  return r;
}

static __device__ __forceinline__ bool mask_ok(int sq, int fq, int nq,
                                               int sk, int fk, int nk, int W) {
  bool t = (nq == 1 && nk == 1 && fk <= fq) ||
           (nq == 0 && nk == 1 && fk <  fq) ||
           (nq == 0 && nk == 0 && fk == fq);
  int df = fq - fk; if (df < 0) df = -df;
  return t && (sq == sk) && (sq >= 0) && (sk >= 0) && (df <= W);
}

// ---------------------------------------------------------------------------
// Kernel A1: K fp32 [s][h][d] -> bf16 Kb [h][s][d] (convert + head-major).
// One thread per bf16 pair; reads/writes coalesced along d.
// ---------------------------------------------------------------------------
__global__ __launch_bounds__(256) void kconv_kernel(const float* __restrict__ k,
                                                    unsigned int* __restrict__ kb) {
  const size_t p = (size_t)blockIdx.x * 256 + threadIdx.x;  // pair index
  const int d2 = (int)(p & 63);            // pair within head_dim (128/2)
  const int h  = (int)((p >> 6) % NH);
  const int s  = (int)(p / (64 * NH));
  const float2 in = ((const float2*)k)[((size_t)s * NH + h) * (HD / 2) + d2];
  unsigned int bits = (unsigned int)f2bf_bits(in.x) |
                      ((unsigned int)f2bf_bits(in.y) << 16);
  kb[((size_t)h * SEQ + s) * (HD / 2) + d2] = bits;
}

// ---------------------------------------------------------------------------
// Kernel A2: transpose+convert V fp32 [s][h][d] -> bf16 Vt [h][d][s]
// so PV B-fragments become contiguous 32B loads per lane.
// ---------------------------------------------------------------------------
__global__ __launch_bounds__(256) void vtrans_kernel(const float* __restrict__ v,
                                                     unsigned short* __restrict__ vt) {
  __shared__ float tile[32][33];
  const int sbase = blockIdx.x * 32;     // 129 tiles * 32 = 4128
  const int dbase = blockIdx.y * 32;     // 4 tiles * 32 = 128
  const int h     = blockIdx.z;
  const int tid   = threadIdx.x;
  const int c0 = tid & 31, r0 = tid >> 5;
#pragma unroll
  for (int i = 0; i < 4; ++i) {
    int s_l = r0 + i * 8;
    tile[s_l][c0] = v[(size_t)(sbase + s_l) * (NH * HD) + h * HD + dbase + c0];
  }
  __syncthreads();
#pragma unroll
  for (int i = 0; i < 4; ++i) {
    int d_l = r0 + i * 8;
    vt[(size_t)(h * HD + dbase + d_l) * SEQ + sbase + c0] = f2bf_bits(tile[c0][d_l]);
  }
}

// ---------------------------------------------------------------------------
// Kernel B: one wave32 per (16-query tile, head). Flash attention over key
// tiles of 32, bf16 WMMA, tile skipping from the block-sparse mask.
// ---------------------------------------------------------------------------
__global__ __launch_bounds__(32) void attn_kernel(
    const float* __restrict__ q, const unsigned short* __restrict__ kb,
    const unsigned short* __restrict__ vt,
    const int* __restrict__ sid, const int* __restrict__ fid,
    const int* __restrict__ nid, const int* __restrict__ wptr,
    unsigned short* __restrict__ out) {
  __shared__ unsigned short Pld[16 * 32];  // bf16 bits, 16 q-rows x 32 keys

  const int lane  = threadIdx.x;   // 0..31
  const int half  = lane >> 4;     // 0/1
  const int l16   = lane & 15;
  const int h     = blockIdx.y;
  const int qbase = blockIdx.x * 16;
  const int W     = wptr[0];
  const float scale = 0.08838834764831845f;  // 1/sqrt(128)

  // ---- Q fragments (A layout), converted once, live for the whole loop ----
  v16bf qa[4];
  {
    const float* qrow = q + ((size_t)(qbase + l16) * NH + h) * HD;
#pragma unroll
    for (int c = 0; c < 4; ++c) {
      const float* p0 = qrow + c * 32 + 16 * half;  // elems 0..7
      const float* p1 = p0 + 8;                     // elems 8..15
      qa[c] = load8p8(p0, p1);
    }
  }

  // ---- per-row (element r) query-side ids ----
  int sq[8], fq[8], nq[8];
#pragma unroll
  for (int r = 0; r < 8; ++r) {
    int qi = qbase + r + 8 * half;   // C-layout row for this lane/element
    sq[r] = sid[qi]; fq[r] = fid[qi]; nq[r] = nid[qi];
  }

  const v8f vzero = {0.f, 0.f, 0.f, 0.f, 0.f, 0.f, 0.f, 0.f};
  float mrow[8], lrow[8];
  v8f o[8];
#pragma unroll
  for (int r = 0; r < 8; ++r) { mrow[r] = -3.0e38f; lrow[r] = 0.f; }
#pragma unroll
  for (int t = 0; t < 8; ++t) o[t] = vzero;

  const uint4* kq  = (const uint4*)kb;   // 8 bf16 per uint4
  const uint4* vtq = (const uint4*)vt;

  for (int kt = 0; kt < SEQ / 32; ++kt) {
    const int kbase = kt * 32;
    const int k0 = kbase + l16;        // key column for S-tile 0 (n = lane%16)
    const int k1 = k0 + 16;            // key column for S-tile 1
    const int sk0 = sid[k0], fk0 = fid[k0], nk0 = nid[k0];
    const int sk1 = sid[k1], fk1 = fid[k1], nk1 = nid[k1];

    unsigned mask0 = 0, mask1 = 0;
#pragma unroll
    for (int r = 0; r < 8; ++r) {
      if (mask_ok(sq[r], fq[r], nq[r], sk0, fk0, nk0, W)) mask0 |= 1u << r;
      if (mask_ok(sq[r], fq[r], nq[r], sk1, fk1, nk1, W)) mask1 |= 1u << r;
    }
    if (!__any((int)(mask0 | mask1))) continue;   // dead tile: skip all work

    // ---- S = Q K^T for 32 keys (2 N-tiles x 4 K-chunks), bf16 K from Kb ----
    v8f s0 = vzero, s1 = vzero;
    // B layout: per-lane 16 contiguous bf16 at d = 32c + 16*half of key row.
    const uint4* kr0 = kq + ((size_t)(h * SEQ + k0) * (HD / 8) + 2 * half);
    const uint4* kr1 = kq + ((size_t)(h * SEQ + k1) * (HD / 8) + 2 * half);
#pragma unroll
    for (int c = 0; c < 4; ++c) {
      union { uint4 u[2]; v16bf v; } b0, b1;
      b0.u[0] = kr0[c * 4 + 0]; b0.u[1] = kr0[c * 4 + 1];
      b1.u[0] = kr1[c * 4 + 0]; b1.u[1] = kr1[c * 4 + 1];
      s0 = wmma_bf16(qa[c], b0.v, s0);
      s1 = wmma_bf16(qa[c], b1.v, s1);
    }

    // ---- masked online softmax (row = element r, spread over a 16-lane half)
    float p0[8], p1[8], alpha[8];
#pragma unroll
    for (int r = 0; r < 8; ++r) {
      float v0 = (mask0 >> r & 1) ? s0[r] * scale : -3.0e38f;
      float v1 = (mask1 >> r & 1) ? s1[r] * scale : -3.0e38f;
      float tmax = fmaxf(v0, v1);
      tmax = fmaxf(tmax, __shfl_xor(tmax, 1));
      tmax = fmaxf(tmax, __shfl_xor(tmax, 2));
      tmax = fmaxf(tmax, __shfl_xor(tmax, 4));
      tmax = fmaxf(tmax, __shfl_xor(tmax, 8));
      float mn = fmaxf(mrow[r], tmax);
      alpha[r] = __expf(mrow[r] - mn);
      mrow[r] = mn;
      p0[r] = (mask0 >> r & 1) ? __expf(v0 - mn) : 0.f;
      p1[r] = (mask1 >> r & 1) ? __expf(v1 - mn) : 0.f;
      float su = p0[r] + p1[r];
      su += __shfl_xor(su, 1);
      su += __shfl_xor(su, 2);
      su += __shfl_xor(su, 4);
      su += __shfl_xor(su, 8);
      lrow[r] = lrow[r] * alpha[r] + su;
    }
#pragma unroll
    for (int t = 0; t < 8; ++t)
#pragma unroll
      for (int r = 0; r < 8; ++r) o[t][r] *= alpha[r];

    // ---- P (C layout) -> LDS -> P (A layout) ----
#pragma unroll
    for (int r = 0; r < 8; ++r) {
      int m = r + 8 * half;
      Pld[m * 32 + l16]      = f2bf_bits(p0[r]);
      Pld[m * 32 + l16 + 16] = f2bf_bits(p1[r]);
    }
    asm volatile("s_wait_dscnt 0" ::: "memory");   // LDS is in-order per wave

    union { uint4 u[2]; v16bf v; } pa;
    const uint4* pl = (const uint4*)Pld;           // 4 uint4 per 32-key row
    pa.u[0] = pl[l16 * 4 + 2 * half];              // keys 16*half .. +7
    pa.u[1] = pl[l16 * 4 + 2 * half + 1];          // keys 16*half+8 .. +7

    // ---- O += P * V  (8 head-dim tiles of 16) ----
#pragma unroll
    for (int t = 0; t < 8; ++t) {
      const uint4* vp = vtq + ((size_t)(h * HD + 16 * t + l16) * (SEQ / 8)
                               + (size_t)(kbase + 16 * half) / 8);
      union { uint4 u[2]; v16bf v; } vb;
      vb.u[0] = vp[0];
      vb.u[1] = vp[1];
      o[t] = wmma_bf16(pa.v, vb.v, o[t]);
    }
    asm volatile("s_wait_dscnt 0" ::: "memory");   // WAR before next P store
  }

  // ---- epilogue: normalize; rows with no valid key -> 0 ----
#pragma unroll
  for (int r = 0; r < 8; ++r) {
    const int m = r + 8 * half;
    const float inv = (lrow[r] > 0.f) ? 1.0f / lrow[r] : 0.f;
    const size_t s = (size_t)(qbase + m);
#pragma unroll
    for (int t = 0; t < 8; ++t) {
      out[(s * NH + h) * HD + 16 * t + l16] = f2bf_bits(o[t][r] * inv);
    }
  }
}

// ---------------------------------------------------------------------------
extern "C" void kernel_launch(void* const* d_in, const int* in_sizes, int n_in,
                              void* d_out, int out_size, void* d_ws, size_t ws_size,
                              hipStream_t stream) {
  const float* q = (const float*)d_in[0];
  const float* k = (const float*)d_in[1];
  const float* v = (const float*)d_in[2];
  const int* sid = (const int*)d_in[3];
  const int* fid = (const int*)d_in[4];
  const int* nid = (const int*)d_in[5];
  const int* w   = (const int*)d_in[6];

  const size_t elems = (size_t)SEQ * NH * HD;           // 6,340,608
  unsigned short* kb = (unsigned short*)d_ws;           // bf16 K:  [12][4128][128]
  unsigned short* vt = kb + elems;                      // bf16 V^T:[12][128][4128]

  kconv_kernel<<<(unsigned)(elems / 2 / 256), 256, 0, stream>>>(
      k, (unsigned int*)kb);
  vtrans_kernel<<<dim3(SEQ / 32, HD / 32, NH), 256, 0, stream>>>(v, vt);
  attn_kernel<<<dim3(SEQ / 16, NH), 32, 0, stream>>>(
      q, kb, vt, sid, fid, nid, w, (unsigned short*)d_out);
}
